// RelationalLayerModel_85409719648962
// MI455X (gfx1250) — compile-verified, matrix operand order checked by reading, not run on hardware
//
#include <hip/hip_runtime.h>
#include <hip/hip_bf16.h>
#include <math.h>

// ---------------------------------------------------------------------------
// Relational Network on gfx1250.
// Layer-1 decomposed (h1 = relu(A[j] + B[i] + Cq)); layers 2-4 are bf16 WMMA
// (v_wmma_f32_16x16x32_bf16) with f32 accumulate, activations resident in LDS,
// W2/W3 in LDS (reused by 1024 rows per workgroup), W4 streamed from L2.
// Two n-tiles in flight with zero-initialized accumulators (bias folded into
// the epilogue to avoid register-pressure spills) and register-prefetched
// B tiles to hide LDS/L2 latency between WMMAs.
// ---------------------------------------------------------------------------

typedef __bf16 bf16_t;
typedef __attribute__((ext_vector_type(16))) __bf16 bf16x16;
typedef __attribute__((ext_vector_type(8)))  __bf16 bf16x8;
typedef __attribute__((ext_vector_type(8)))  float  f32x8;

struct Pair128 { int4 lo; int4 hi; };   // 32 bytes == bf16x16

#define WMMA_BF16(A, B, C)                                                     \
  __builtin_amdgcn_wmma_f32_16x16x32_bf16(false, (A), false, (B), (short)0,    \
                                          (C), false, false)

// --------------------------- setup kernels ---------------------------------

// Af[b,o,h] = xf[b,o,:26] @ g1_w[0:26], Bf = xf @ g1_w[26:52]
__global__ __launch_bounds__(256) void rn_prep_obj(
    const float* __restrict__ x, const float* __restrict__ g1w,
    float* __restrict__ Af, float* __restrict__ Bf) {
  __shared__ float xf[26];
  int blk = blockIdx.x;
  int b = blk >> 6, o = blk & 63;
  int h = threadIdx.x;
  if (h < 26) {
    float v;
    if (h < 24)      v = x[(b * 24 + h) * 64 + o];
    else if (h == 24) v = ((float)o * 0.125f - 4.0f) * 0.25f;       // cx (float div!)
    else              v = ((float)(o & 7) - 4.0f) * 0.25f;          // cy
    xf[h] = v;
  }
  __syncthreads();
  float a = 0.f, bb = 0.f;
  for (int c = 0; c < 26; ++c) {
    float xv = xf[c];
    a  += xv * g1w[c * 256 + h];
    bb += xv * g1w[(26 + c) * 256 + h];
  }
  Af[(b * 64 + o) * 256 + h] = a;
  Bf[(b * 64 + o) * 256 + h] = bb;
}

// Cq[b,h] = qst[b] @ g1_w[52:180] + g1_b ; also zero xg (re-zeroed every call)
__global__ __launch_bounds__(256) void rn_prep_q(
    const float* __restrict__ qst, const float* __restrict__ g1w,
    const float* __restrict__ g1b, float* __restrict__ Cq,
    float* __restrict__ xg) {
  int b = blockIdx.x, h = threadIdx.x;
  float acc = g1b[h];
  for (int q = 0; q < 128; ++q)
    acc += qst[b * 128 + q] * g1w[(52 + q) * 256 + h];
  Cq[b * 256 + h] = acc;
  xg[b * 256 + h] = 0.f;
}

// Swizzle g2/g3/g4 (f32 [K=256][N=256]) into bf16 WMMA B-operand order:
// [kt(8)][nt(16)][lane(32)][16 bf16], lane = half*16+idx, col n = nt*16+idx,
// elems 0..7 -> k = kt*32 + half*8 + e ; elems 8..15 -> k = kt*32+16+half*8+(e-8)
__global__ __launch_bounds__(256) void rn_swz(
    const float* __restrict__ w2, const float* __restrict__ w3,
    const float* __restrict__ w4, bf16_t* __restrict__ dst) {
  int t = blockIdx.x * 256 + threadIdx.x;         // 0 .. 3*65536-1
  int layer = t >> 16;
  int L = t & 65535;
  int e    = L & 15;
  int lane = (L >> 4) & 31;
  int nt   = (L >> 9) & 15;
  int kt   = L >> 13;
  int half = lane >> 4, idx = lane & 15;
  int k = kt * 32 + half * 8 + (e < 8 ? e : 16 + (e - 8));
  int n = nt * 16 + idx;
  const float* src = (layer == 0) ? w2 : (layer == 1 ? w3 : w4);
  dst[t] = (bf16_t)src[k * 256 + n];
}

// --------------------------- main WMMA kernel ------------------------------

// Stage layout: logical (row, c) stored at stg[row*256 + ((c + row*8)&255)]
// (8-element rotation per row -> conflict-free ds_load_b128 A-tile reads).
__device__ __forceinline__ bf16x16 ld_stage_tile(const bf16_t* stg, int idx,
                                                 int half, int kt) {
  int kbase = kt * 32 + half * 8;
  int c0 = (kbase + idx * 8) & 255;          // k = kbase .. kbase+7
  int c1 = (kbase + 16 + idx * 8) & 255;     // k = kbase+16 .. kbase+23
  Pair128 r;
  r.lo = *(const int4*)(stg + idx * 256 + c0);
  r.hi = *(const int4*)(stg + idx * 256 + c1);
  return __builtin_bit_cast(bf16x16, r);
}

__device__ __forceinline__ bf16x16 ld_w_tile(const bf16_t* W, int kt, int nt,
                                             int lane) {
  const bf16_t* p = W + (((kt * 16 + nt) * 32 + lane) << 4);
  Pair128 r;
  r.lo = ((const int4*)p)[0];
  r.hi = ((const int4*)p)[1];
  return __builtin_bit_cast(bf16x16, r);
}

// one 16x256 activation block through relu(X @ W + bias), written back in-place
// two n-tiles in flight, B tiles prefetched one k-step ahead (LDS weights);
// accumulators start at zero, bias folded into the relu epilogue.
__device__ __forceinline__ void layer_pass(bf16_t* stg, const bf16_t* W,
                                           const float* __restrict__ bias,
                                           int lane) {
  int half = lane >> 4, idx = lane & 15;
  bf16x16 a[8];
#pragma unroll
  for (int kt = 0; kt < 8; ++kt) a[kt] = ld_stage_tile(stg, idx, half, kt);
#pragma unroll
  for (int nt = 0; nt < 16; nt += 2) {
    f32x8 acc0 = {};
    f32x8 acc1 = {};
    bf16x16 b0 = ld_w_tile(W, 0, nt, lane);
    bf16x16 b1 = ld_w_tile(W, 0, nt + 1, lane);
#pragma unroll
    for (int kt = 0; kt < 8; ++kt) {
      bf16x16 n0, n1;
      if (kt < 7) {
        n0 = ld_w_tile(W, kt + 1, nt, lane);
        n1 = ld_w_tile(W, kt + 1, nt + 1, lane);
      }
      acc0 = WMMA_BF16(a[kt], b0, acc0);
      acc1 = WMMA_BF16(a[kt], b1, acc1);
      if (kt < 7) { b0 = n0; b1 = n1; }
    }
    float bv0 = bias[nt * 16 + idx];
    float bv1 = bias[nt * 16 + 16 + idx];
#pragma unroll
    for (int v = 0; v < 8; ++v) {
      int row = v + half * 8;                        // C layout row
      int col0 = (nt * 16 + idx + row * 8) & 255;
      int col1 = (nt * 16 + 16 + idx + row * 8) & 255;
      stg[row * 256 + col0] = (bf16_t)fmaxf(acc0[v] + bv0, 0.f);
      stg[row * 256 + col1] = (bf16_t)fmaxf(acc1[v] + bv1, 0.f);
    }
  }
}

// last g layer: relu then reduce rows into per-column partial sums.
// W4 streamed from global/L2 -> depth-2 register double-buffer prefetch.
__device__ __forceinline__ void layer4_pass(const bf16_t* stg,
                                            const bf16_t* W4g,
                                            const float* __restrict__ bias,
                                            int lane, float* colsum) {
  int half = lane >> 4, idx = lane & 15;
  bf16x16 a[8];
#pragma unroll
  for (int kt = 0; kt < 8; ++kt) a[kt] = ld_stage_tile(stg, idx, half, kt);
#pragma unroll
  for (int nt = 0; nt < 16; nt += 2) {
    f32x8 acc0 = {};
    f32x8 acc1 = {};
    bf16x16 buf0[2], buf1[2];
    buf0[0] = ld_w_tile(W4g, 0, nt, lane);
    buf1[0] = ld_w_tile(W4g, 0, nt + 1, lane);
    buf0[1] = ld_w_tile(W4g, 1, nt, lane);
    buf1[1] = ld_w_tile(W4g, 1, nt + 1, lane);
#pragma unroll
    for (int kt = 0; kt < 8; ++kt) {
      bf16x16 c0 = buf0[kt & 1];
      bf16x16 c1 = buf1[kt & 1];
      if (kt < 6) {
        buf0[kt & 1] = ld_w_tile(W4g, kt + 2, nt, lane);
        buf1[kt & 1] = ld_w_tile(W4g, kt + 2, nt + 1, lane);
      }
      acc0 = WMMA_BF16(a[kt], c0, acc0);
      acc1 = WMMA_BF16(a[kt], c1, acc1);
    }
    float bv0 = bias[nt * 16 + idx];
    float bv1 = bias[nt * 16 + 16 + idx];
    float s0 = 0.f, s1 = 0.f;
#pragma unroll
    for (int v = 0; v < 8; ++v) {
      s0 += fmaxf(acc0[v] + bv0, 0.f);
      s1 += fmaxf(acc1[v] + bv1, 0.f);
    }
    colsum[nt] += s0;       // lane covers rows half*8..half*8+7 of col nt*16+idx
    colsum[nt + 1] += s1;
  }
}

__global__ __launch_bounds__(256, 1) void rn_main(
    const float* __restrict__ Af, const float* __restrict__ Bf,
    const float* __restrict__ Cq, const bf16_t* __restrict__ Wswz,
    const float* __restrict__ b2, const float* __restrict__ b3,
    const float* __restrict__ b4, float* __restrict__ xg) {
  __shared__ __align__(16) bf16_t sW2[65536];       // 128 KB
  __shared__ __align__(16) bf16_t sW3[65536];       // 128 KB
  __shared__ __align__(16) bf16_t stage[8][4096];   //  64 KB (8 KB per wave)

  int tid = threadIdx.x;
  int wave = tid >> 5, lane = tid & 31;
  int b = blockIdx.x >> 2, iblk = blockIdx.x & 3;

  // cooperative load of W2/W3 into LDS (b128 copies)
  {
    const int4* s2 = (const int4*)Wswz;
    const int4* s3 = (const int4*)(Wswz + 65536);
    int4* d2 = (int4*)sW2;
    int4* d3 = (int4*)sW3;
    for (int t = tid; t < 8192; t += 256) { d2[t] = s2[t]; d3[t] = s3[t]; }
  }
  __syncthreads();

  const bf16_t* W4g = Wswz + 2 * 65536;
  bf16_t* stg = stage[wave];
  float colsum[16];
#pragma unroll
  for (int nt = 0; nt < 16; ++nt) colsum[nt] = 0.f;

  // 1024 rows per workgroup = 64 chunks of 16 rows; 8 chunks per wave.
  for (int c = 0; c < 8; ++c) {
    int ch = wave * 8 + c;
    int i_g = iblk * 16 + (ch >> 2);   // fixed i within chunk
    int j0 = (ch & 3) * 16;            // 16 consecutive j

    // ---- build h1 = relu(A[j] + B[i] + Cq) into stage (bf16) ----
    const float4* bp = (const float4*)(Bf + ((b * 64 + i_g) << 8) + (lane << 3));
    const float4* cp = (const float4*)(Cq + (b << 8) + (lane << 3));
    float4 bv0 = bp[0], bv1 = bp[1];
    float4 cv0 = cp[0], cv1 = cp[1];
    float bc0 = bv0.x + cv0.x, bc1 = bv0.y + cv0.y;
    float bc2 = bv0.z + cv0.z, bc3 = bv0.w + cv0.w;
    float bc4 = bv1.x + cv1.x, bc5 = bv1.y + cv1.y;
    float bc6 = bv1.z + cv1.z, bc7 = bv1.w + cv1.w;
    for (int rr = 0; rr < 16; ++rr) {
      const float4* ap =
          (const float4*)(Af + ((b * 64 + j0 + rr) << 8) + (lane << 3));
      float4 a0 = ap[0], a1 = ap[1];
      bf16x8 pk;
      pk[0] = (bf16_t)fmaxf(a0.x + bc0, 0.f);
      pk[1] = (bf16_t)fmaxf(a0.y + bc1, 0.f);
      pk[2] = (bf16_t)fmaxf(a0.z + bc2, 0.f);
      pk[3] = (bf16_t)fmaxf(a0.w + bc3, 0.f);
      pk[4] = (bf16_t)fmaxf(a1.x + bc4, 0.f);
      pk[5] = (bf16_t)fmaxf(a1.y + bc5, 0.f);
      pk[6] = (bf16_t)fmaxf(a1.z + bc6, 0.f);
      pk[7] = (bf16_t)fmaxf(a1.w + bc7, 0.f);
      int col = ((lane << 3) + (rr << 3)) & 255;    // rotated store
      *(int4*)(stg + rr * 256 + col) = __builtin_bit_cast(int4, pk);
    }

    // ---- layers 2..4 ----
    layer_pass(stg, sW2, b2, lane);
    layer_pass(stg, sW3, b3, lane);
    layer4_pass(stg, W4g, b4, lane, colsum);
  }

  int idx = lane & 15;
#pragma unroll
  for (int nt = 0; nt < 16; ++nt)
    atomicAdd(&xg[(b << 8) + nt * 16 + idx], colsum[nt]);
}

// --------------------------- f-MLP + log_softmax ---------------------------

__global__ __launch_bounds__(256) void rn_fmlp(
    const float* __restrict__ xg, const float* __restrict__ f1w,
    const float* __restrict__ f1b, const float* __restrict__ f2w,
    const float* __restrict__ f2b, const float* __restrict__ f3w,
    const float* __restrict__ f3b, float* __restrict__ out) {
  __shared__ float t0[256], t1[256], lg[28], red[2];
  int b = blockIdx.x, h = threadIdx.x;
  t0[h] = xg[b * 256 + h];
  __syncthreads();
  float acc = f1b[h];
  for (int c = 0; c < 256; ++c) acc += t0[c] * f1w[c * 256 + h];
  t1[h] = fmaxf(acc, 0.f);
  __syncthreads();
  acc = f2b[h];
  for (int c = 0; c < 256; ++c) acc += t1[c] * f2w[c * 256 + h];
  t0[h] = fmaxf(acc, 0.f);
  __syncthreads();
  if (h < 28) {
    float a = f3b[h];
    for (int c = 0; c < 256; ++c) a += t0[c] * f3w[c * 28 + h];
    lg[h] = a;
  }
  __syncthreads();
  if (h == 0) {
    float m = lg[0];
    for (int i = 1; i < 28; ++i) m = fmaxf(m, lg[i]);
    float s = 0.f;
    for (int i = 0; i < 28; ++i) s += expf(lg[i] - m);
    red[0] = m;
    red[1] = logf(s);
  }
  __syncthreads();
  if (h < 28) out[b * 28 + h] = lg[h] - red[0] - red[1];
}

// --------------------------- launcher --------------------------------------

extern "C" void kernel_launch(void* const* d_in, const int* in_sizes, int n_in,
                              void* d_out, int out_size, void* d_ws,
                              size_t ws_size, hipStream_t stream) {
  const float* x   = (const float*)d_in[0];
  const float* qst = (const float*)d_in[1];
  const float* g1w = (const float*)d_in[2];
  const float* g1b = (const float*)d_in[3];
  const float* g2w = (const float*)d_in[4];
  const float* g2b = (const float*)d_in[5];
  const float* g3w = (const float*)d_in[6];
  const float* g3b = (const float*)d_in[7];
  const float* g4w = (const float*)d_in[8];
  const float* g4b = (const float*)d_in[9];
  const float* f1w = (const float*)d_in[10];
  const float* f1b = (const float*)d_in[11];
  const float* f2w = (const float*)d_in[12];
  const float* f2b = (const float*)d_in[13];
  const float* f3w = (const float*)d_in[14];
  const float* f3b = (const float*)d_in[15];

  float* Af = (float*)d_ws;                 // 64*64*256 f32  (4 MB)
  float* Bf = Af + 64 * 64 * 256;           // 4 MB
  float* Cq = Bf + 64 * 64 * 256;           // 64 KB
  float* xg = Cq + 64 * 256;                // 64 KB
  bf16_t* Wswz = (bf16_t*)(xg + 64 * 256);  // 3 * 65536 bf16 (384 KB)

  rn_prep_obj<<<64 * 64, 256, 0, stream>>>(x, g1w, Af, Bf);
  rn_prep_q<<<64, 256, 0, stream>>>(qst, g1w, g1b, Cq, xg);
  rn_swz<<<768, 256, 0, stream>>>(g2w, g3w, g4w, Wswz);
  rn_main<<<256, 256, 0, stream>>>(Af, Bf, Cq, Wswz, g2b, g3b, g4b, xg);
  rn_fmlp<<<64, 256, 0, stream>>>(xg, f1w, f1b, f2w, f2b, f3w, f3b,
                                  (float*)d_out);
}
